// MLPBlock_34479997452739
// MI455X (gfx1250) — matrix operand order repaired and checked by reading, not compiled, not run on hardware
//
#include <hip/hip_runtime.h>
#include <hip/hip_bf16.h>
#include <math.h>

// ---- problem constants (match reference) ----
#define T_TOK 1024
#define H_DIM 2880
#define I_DIM 2880
#define N2    (2 * I_DIM)   // 5760
#define E_NUM 32
#define K_TOP 4
#define CAP   384
#define LIMITF 7.0f
#define ALPHAF 1.702f
#define EPSF   1e-5f

#define TILE_SHORTS (96 * 32)        // ushorts per LDS buffer
#define TILE_DWORDS (96 * 16)        // dwords per LDS buffer

typedef __attribute__((ext_vector_type(16))) __bf16 v16bf;
typedef __attribute__((ext_vector_type(8)))  float  v8f;

union FragU {
    uint4          q[2];
    unsigned short s[16];
    v16bf          v;
};
union PackU {
    uint4          q;
    unsigned short s[8];
};

// ---------------- Kernel 1: RMSNorm (fp32 accum) ----------------
__global__ void k_rmsnorm(const __hip_bfloat16* __restrict__ x,
                          const float* __restrict__ scale,
                          __hip_bfloat16* __restrict__ t) {
    int row = blockIdx.x, tid = threadIdx.x;
    __shared__ float red[256];
    const __hip_bfloat16* xr = x + (size_t)row * H_DIM;
    float ss = 0.f;
    for (int h = tid; h < H_DIM; h += 256) {
        float v = __bfloat162float(xr[h]);
        ss += v * v;
    }
    red[tid] = ss; __syncthreads();
    for (int s = 128; s > 0; s >>= 1) {
        if (tid < s) red[tid] += red[tid + s];
        __syncthreads();
    }
    float r = rsqrtf(red[0] / (float)H_DIM + EPSF);
    __hip_bfloat16* tr = t + (size_t)row * H_DIM;
    for (int h = tid; h < H_DIM; h += 256) {
        float v = __bfloat162float(xr[h]);
        tr[h] = __float2bfloat16(scale[h] * v * r);
    }
}

// ---------------- Kernel 2: router gemv + top-4 + softmax ----------------
__global__ void k_router(const __hip_bfloat16* __restrict__ t,
                         const __hip_bfloat16* __restrict__ gw,
                         const __hip_bfloat16* __restrict__ gb,
                         int* __restrict__ top_idx,
                         float* __restrict__ top_w) {
    int row = blockIdx.x;
    int lane = threadIdx.x;            // 32 lanes == 32 experts
    const __hip_bfloat16* tr = t + (size_t)row * H_DIM;
    float acc = __bfloat162float(gb[lane]);
    for (int h = 0; h < H_DIM; ++h)
        acc += __bfloat162float(tr[h]) * __bfloat162float(gw[(size_t)h * E_NUM + lane]);
    float v = acc;
    float tv[K_TOP]; int ti[K_TOP];
    for (int k = 0; k < K_TOP; ++k) {
        float m = v; int mi = lane;
        for (int off = 16; off > 0; off >>= 1) {
            float om = __shfl_xor(m, off, 32);
            int  omi = __shfl_xor(mi, off, 32);
            if (om > m || (om == m && omi < mi)) { m = om; mi = omi; }
        }
        tv[k] = m; ti[k] = mi;
        if (lane == mi) v = -3.0e38f;
    }
    if (lane == 0) {
        float mx = tv[0];
        float e0 = __expf(tv[0] - mx), e1 = __expf(tv[1] - mx);
        float e2 = __expf(tv[2] - mx), e3 = __expf(tv[3] - mx);
        float inv = 1.f / (e0 + e1 + e2 + e3);
        top_idx[row * 4 + 0] = ti[0]; top_w[row * 4 + 0] = e0 * inv;
        top_idx[row * 4 + 1] = ti[1]; top_w[row * 4 + 1] = e1 * inv;
        top_idx[row * 4 + 2] = ti[2]; top_w[row * 4 + 2] = e2 * inv;
        top_idx[row * 4 + 3] = ti[3]; top_w[row * 4 + 3] = e3 * inv;
    }
}

// ------ Kernel 3: deterministic capacity dispatch (stable order, ballot scan) ------
__global__ void k_dispatch(const int* __restrict__ top_idx,
                           const float* __restrict__ top_w,
                           int* __restrict__ slot_of,
                           int* __restrict__ disp_tok,
                           float* __restrict__ disp_w,
                           int* __restrict__ counts) {
    int e = blockIdx.x;
    int lane = threadIdx.x;            // 32 lanes
    unsigned ltmask = (1u << lane) - 1u;
    int base = 0;
    const int N = T_TOK * K_TOP;       // multiple of 32
    for (int i = 0; i < N; i += 32) {
        int idx = i + lane;
        bool match = (top_idx[idx] == e);
        unsigned m = (unsigned)__ballot(match);
        if (match) {
            int pos = base + __popc(m & ltmask);
            if (pos < CAP) {
                disp_tok[e * CAP + pos] = idx >> 2;     // token id
                disp_w[e * CAP + pos]   = top_w[idx];
                slot_of[idx] = e * CAP + pos;           // global slot
            } else {
                slot_of[idx] = -1;                      // dropped
            }
        }
        base += __popc(m);
    }
    int cnt = base < CAP ? base : CAP;
    if (lane == 0) counts[e] = cnt;
    for (int p = cnt + lane; p < CAP; p += 32) {        // pad slots: token 0, weight 0
        disp_tok[e * CAP + p] = 0;
        disp_w[e * CAP + p]   = 0.f;
    }
}

// ============================================================================
// GEMM tile engine: block = 6 waves (192 thr), tile 64(M) x 96(N), K-step 32.
// Double-buffered LDS B tile (column-major), packed-dword staging.
// Each wave: 1 B fragment, 4 A fragments, 4 WMMAs per K-step.
// ============================================================================

// ------ Kernel 4: expert GEMM1 (xe @ W1 + b1) fused with interleaved SwiGLU ------
__global__ __launch_bounds__(192) void k_gemm1_swiglu(
        const __hip_bfloat16* __restrict__ t,
        const __hip_bfloat16* __restrict__ W1,
        const __hip_bfloat16* __restrict__ b1,
        const int* __restrict__ disp_tok,
        const int* __restrict__ counts,
        __hip_bfloat16* __restrict__ act) {
    int e  = blockIdx.z;
    int m0 = blockIdx.y * 64;
    if (m0 >= counts[e]) return;       // skip empty capacity tiles (uniform)
    int n0 = blockIdx.x * 96;
    int tid = threadIdx.x;
    int wv = tid >> 5, lane = tid & 31, r = lane & 15;
    int kb = (lane & 16) ? 8 : 0;      // unused; A fragment is offset by kh like B
    int kh = (lane & 16) ? 16 : 0;     // B-fragment K half
    __shared__ unsigned short ldsB[2 * TILE_SHORTS];
    unsigned* ldsW = (unsigned*)ldsB;  // dword view for packed staging

    const unsigned short* tp = (const unsigned short*)t;
    const unsigned short* w1 = (const unsigned short*)W1 + (size_t)e * H_DIM * N2;

    // 4 gathered activation rows per lane (one per M sub-tile)
    const unsigned short* arow[4];
    #pragma unroll
    for (int s = 0; s < 4; ++s) {
        int tok = disp_tok[e * CAP + m0 + s * 16 + r];
        arow[s] = tp + (size_t)tok * H_DIM;
    }

    // staging map: 16 k-pairs x 12 n-chunks of 8
    int kp = tid / 12;                 // 0..15 -> rows 2kp, 2kp+1
    int nn = (tid % 12) * 8;           // 0..88

    // prologue: stage tile 0
    PackU c0, c1;
    {
        const unsigned short* gp = w1 + (size_t)(2 * kp) * N2 + n0 + nn;
        c0.q = *(const uint4*)gp;
        c1.q = *(const uint4*)(gp + N2);
        #pragma unroll
        for (int j = 0; j < 8; ++j)
            ldsW[(nn + j) * 16 + kp] = (unsigned)c0.s[j] | ((unsigned)c1.s[j] << 16);
    }
    __syncthreads();

    v8f acc[4] = {};
    for (int k0 = 0; k0 < H_DIM; k0 += 32) {
        int buf = (k0 >> 5) & 1;
        bool more = (k0 + 32 < H_DIM);
        PackU n0q, n1q;
        if (more) {
            const unsigned short* gp = w1 + (size_t)(k0 + 32 + 2 * kp) * N2 + n0 + nn;
            n0q.q = *(const uint4*)gp;
            n1q.q = *(const uint4*)(gp + N2);
            if (k0 + 64 < H_DIM) __builtin_prefetch(gp + (size_t)32 * N2, 0, 1);
        }
        // B fragment from LDS (column-major, contiguous K)
        FragU bf;
        const unsigned short* bp = &ldsB[buf * TILE_SHORTS + (wv * 16 + r) * 32 + kh];
        bf.q[0] = *(const uint4*)bp;
        bf.q[1] = *(const uint4*)(bp + 8);
        // 4 M sub-tiles
        #pragma unroll
        for (int s = 0; s < 4; ++s) {
            FragU a;
            a.q[0] = *(const uint4*)(arow[s] + k0 + kh);
            a.q[1] = *(const uint4*)(arow[s] + k0 + kh + 8);
            acc[s] = __builtin_amdgcn_wmma_f32_16x16x32_bf16(false, a.v, false, bf.v,
                                                             (short)0, acc[s], false, false);
        }
        if (more) {
            int dst = (buf ^ 1) * TILE_DWORDS;
            #pragma unroll
            for (int j = 0; j < 8; ++j)
                ldsW[dst + (nn + j) * 16 + kp] =
                    (unsigned)n0q.s[j] | ((unsigned)n1q.s[j] << 16);
        }
        __syncthreads();
    }

    // epilogue: bias + SwiGLU over interleaved (even=glu, odd=linear) columns
    int col = n0 + wv * 16 + r;
    float bias = __bfloat162float(b1[(size_t)e * N2 + col]);
    #pragma unroll
    for (int s = 0; s < 4; ++s) {
        int mbase = m0 + s * 16 + ((lane & 16) ? 8 : 0);
        #pragma unroll
        for (int j = 0; j < 8; ++j) {
            float h = acc[s][j] + bias;
            float p = __shfl_xor(h, 1, 32);          // partner column (col ^ 1)
            if ((col & 1) == 0) {
                float xg = fminf(h, LIMITF);
                float xl = fminf(fmaxf(p, -LIMITF), LIMITF);
                float g  = xg / (1.f + __expf(-ALPHAF * xg));   // xg*sigmoid(alpha*xg)
                float av = g * (xl + 1.f);
                act[((size_t)e * CAP + mbase + j) * I_DIM + (col >> 1)] =
                    __float2bfloat16(av);
            }
        }
    }
}

// ------ Kernel 5: expert GEMM2 (act @ W2 + b2) * router weight ------
__global__ __launch_bounds__(192) void k_gemm2(
        const __hip_bfloat16* __restrict__ act,
        const __hip_bfloat16* __restrict__ W2,
        const __hip_bfloat16* __restrict__ b2,
        const float* __restrict__ disp_w,
        const int* __restrict__ counts,
        __hip_bfloat16* __restrict__ o) {
    int e  = blockIdx.z;
    int m0 = blockIdx.y * 64;
    if (m0 >= counts[e]) return;
    int n0 = blockIdx.x * 96;
    int tid = threadIdx.x;
    int wv = tid >> 5, lane = tid & 31, r = lane & 15;
    int kb = (lane & 16) ? 8 : 0;
    int kh = (lane & 16) ? 16 : 0;
    __shared__ unsigned short ldsB[2 * TILE_SHORTS];
    unsigned* ldsW = (unsigned*)ldsB;

    const unsigned short* w2 = (const unsigned short*)W2 + (size_t)e * I_DIM * H_DIM;
    const unsigned short* arow[4];
    #pragma unroll
    for (int s = 0; s < 4; ++s)
        arow[s] = (const unsigned short*)act + ((size_t)e * CAP + m0 + s * 16 + r) * I_DIM;

    int kp = tid / 12;
    int nn = (tid % 12) * 8;

    PackU c0, c1;
    {
        const unsigned short* gp = w2 + (size_t)(2 * kp) * H_DIM + n0 + nn;
        c0.q = *(const uint4*)gp;
        c1.q = *(const uint4*)(gp + H_DIM);
        #pragma unroll
        for (int j = 0; j < 8; ++j)
            ldsW[(nn + j) * 16 + kp] = (unsigned)c0.s[j] | ((unsigned)c1.s[j] << 16);
    }
    __syncthreads();

    v8f acc[4] = {};
    for (int k0 = 0; k0 < I_DIM; k0 += 32) {
        int buf = (k0 >> 5) & 1;
        bool more = (k0 + 32 < I_DIM);
        PackU n0q, n1q;
        if (more) {
            const unsigned short* gp = w2 + (size_t)(k0 + 32 + 2 * kp) * H_DIM + n0 + nn;
            n0q.q = *(const uint4*)gp;
            n1q.q = *(const uint4*)(gp + H_DIM);
            if (k0 + 64 < I_DIM) __builtin_prefetch(gp + (size_t)32 * H_DIM, 0, 1);
        }
        FragU bf;
        const unsigned short* bp = &ldsB[buf * TILE_SHORTS + (wv * 16 + r) * 32 + kh];
        bf.q[0] = *(const uint4*)bp;
        bf.q[1] = *(const uint4*)(bp + 8);
        #pragma unroll
        for (int s = 0; s < 4; ++s) {
            FragU a;
            a.q[0] = *(const uint4*)(arow[s] + k0 + kh);
            a.q[1] = *(const uint4*)(arow[s] + k0 + kh + 8);
            acc[s] = __builtin_amdgcn_wmma_f32_16x16x32_bf16(false, a.v, false, bf.v,
                                                             (short)0, acc[s], false, false);
        }
        if (more) {
            int dst = (buf ^ 1) * TILE_DWORDS;
            #pragma unroll
            for (int j = 0; j < 8; ++j)
                ldsW[dst + (nn + j) * 16 + kp] =
                    (unsigned)n0q.s[j] | ((unsigned)n1q.s[j] << 16);
        }
        __syncthreads();
    }

    int col = n0 + wv * 16 + r;
    float bias = __bfloat162float(b2[(size_t)e * H_DIM + col]);
    #pragma unroll
    for (int s = 0; s < 4; ++s) {
        int mbase = m0 + s * 16 + ((lane & 16) ? 8 : 0);
        #pragma unroll
        for (int j = 0; j < 8; ++j) {
            float wgt = disp_w[e * CAP + mbase + j];
            float val = (acc[s][j] + bias) * wgt;
            o[((size_t)e * CAP + mbase + j) * H_DIM + col] = __float2bfloat16(val);
        }
    }
}

// ------ Kernel 6: deterministic combine (gather 4 slots) + residual ------
__global__ void k_combine(const __hip_bfloat16* __restrict__ x,
                          const __hip_bfloat16* __restrict__ o,
                          const int* __restrict__ slot_of,
                          __hip_bfloat16* __restrict__ out) {
    int row = blockIdx.x, tid = threadIdx.x;
    int s0 = slot_of[row * 4 + 0], s1 = slot_of[row * 4 + 1];
    int s2 = slot_of[row * 4 + 2], s3 = slot_of[row * 4 + 3];
    const __hip_bfloat16* xr = x + (size_t)row * H_DIM;
    __hip_bfloat16* orow = out + (size_t)row * H_DIM;
    for (int h = tid; h < H_DIM; h += 256) {
        float s = __bfloat162float(xr[h]);
        if (s0 >= 0) s += __bfloat162float(o[(size_t)s0 * H_DIM + h]);
        if (s1 >= 0) s += __bfloat162float(o[(size_t)s1 * H_DIM + h]);
        if (s2 >= 0) s += __bfloat162float(o[(size_t)s2 * H_DIM + h]);
        if (s3 >= 0) s += __bfloat162float(o[(size_t)s3 * H_DIM + h]);
        orow[h] = __float2bfloat16(s);
    }
}

extern "C" void kernel_launch(void* const* d_in, const int* in_sizes, int n_in,
                              void* d_out, int out_size, void* d_ws, size_t ws_size,
                              hipStream_t stream) {
    (void)in_sizes; (void)n_in; (void)out_size; (void)ws_size;
    const __hip_bfloat16* x   = (const __hip_bfloat16*)d_in[0];
    const float* norm_scale   = (const float*)d_in[1];
    const __hip_bfloat16* gw  = (const __hip_bfloat16*)d_in[2];
    const __hip_bfloat16* gb  = (const __hip_bfloat16*)d_in[3];
    const __hip_bfloat16* W1  = (const __hip_bfloat16*)d_in[4];
    const __hip_bfloat16* b1  = (const __hip_bfloat16*)d_in[5];
    const __hip_bfloat16* W2  = (const __hip_bfloat16*)d_in[6];
    const __hip_bfloat16* b2  = (const __hip_bfloat16*)d_in[7];
    __hip_bfloat16* out = (__hip_bfloat16*)d_out;

    // workspace carve-up (256B aligned regions, ~148 MB total)
    char* ws = (char*)d_ws;
    size_t off = 0;
    auto carve = [&](size_t bytes) -> void* {
        void* p = ws + off;
        off = (off + bytes + 255) & ~(size_t)255;
        return p;
    };
    __hip_bfloat16* t    = (__hip_bfloat16*)carve((size_t)T_TOK * H_DIM * 2);
    int*   top_idx  = (int*)  carve((size_t)T_TOK * K_TOP * 4);
    float* top_w    = (float*)carve((size_t)T_TOK * K_TOP * 4);
    int*   slot_of  = (int*)  carve((size_t)T_TOK * K_TOP * 4);
    int*   disp_tok = (int*)  carve((size_t)E_NUM * CAP * 4);
    float* disp_w   = (float*)carve((size_t)E_NUM * CAP * 4);
    int*   counts   = (int*)  carve((size_t)E_NUM * 4);
    __hip_bfloat16* actb = (__hip_bfloat16*)carve((size_t)E_NUM * CAP * I_DIM * 2);
    __hip_bfloat16* ob   = (__hip_bfloat16*)carve((size_t)E_NUM * CAP * H_DIM * 2);

    k_rmsnorm <<<T_TOK, 256, 0, stream>>>(x, norm_scale, t);
    k_router  <<<T_TOK, 32, 0, stream>>>(t, gw, gb, top_idx, top_w);
    k_dispatch<<<E_NUM, 32, 0, stream>>>(top_idx, top_w, slot_of, disp_tok, disp_w, counts);
    k_gemm1_swiglu<<<dim3(N2 / 96, CAP / 64, E_NUM), 192, 0, stream>>>(
        t, W1, b1, disp_tok, counts, actb);
    k_gemm2<<<dim3(H_DIM / 96, CAP / 64, E_NUM), 192, 0, stream>>>(
        actb, W2, b2, disp_w, counts, ob);
    k_combine<<<T_TOK, 256, 0, stream>>>(x, ob, slot_of, out);
}
